// MultiHeadAttention_45896020525379
// MI455X (gfx1250) — matrix operand order repaired and checked by reading, not compiled, not run on hardware
//
#include <hip/hip_runtime.h>
#include <hip/hip_bf16.h>

#define EMB   768
#define NSEQ  2048
#define BATCH 4
#define NHEAD 8
#define HDIM  96
#define SROWS (BATCH * NSEQ)   // 8192

typedef __attribute__((ext_vector_type(16))) _Float16 v16h;
typedef __attribute__((ext_vector_type(8)))  float    v8f;

// ---------- helpers -------------------------------------------------------

__device__ __forceinline__ float red_max16(float v) {
    v = fmaxf(v, __shfl_xor(v, 1, 16));
    v = fmaxf(v, __shfl_xor(v, 2, 16));
    v = fmaxf(v, __shfl_xor(v, 4, 16));
    v = fmaxf(v, __shfl_xor(v, 8, 16));
    return v;
}
__device__ __forceinline__ float red_sum16(float v) {
    v += __shfl_xor(v, 1, 16);
    v += __shfl_xor(v, 2, 16);
    v += __shfl_xor(v, 4, 16);
    v += __shfl_xor(v, 8, 16);
    return v;
}

// ---------- fp32 -> f16 conversion ---------------------------------------

__global__ __launch_bounds__(256) void cvt_f32_f16(const float* __restrict__ in,
                                                   _Float16* __restrict__ out, int n) {
    int i = blockIdx.x * blockDim.x + threadIdx.x;
    if (i < n) out[i] = (_Float16)in[i];
}

// transpose-convert a weight matrix: out[n*EMB + k] = (f16) in[k*EMB + n]
// (coalesced writes; strided reads hit L2 / are tiny)
__global__ __launch_bounds__(256) void cvt_w_f16t(const float* __restrict__ in,
                                                  _Float16* __restrict__ out) {
    int idx = blockIdx.x * blockDim.x + threadIdx.x;
    if (idx < EMB * EMB) {
        int n = idx / EMB, k = idx % EMB;
        out[idx] = (_Float16)in[(size_t)k * EMB + n];
    }
}

// ---------- GEMM: C[S x E] = A[S x E] * W[E x E] + bias -------------------
// W is stored TRANSPOSED (Wt[n][k]) so both A and B fragment loads are
// contiguous b128 loads. One wave computes a 16x64 C tile (4 WMMAs / K-step).
// Fragment layouts used here (wave32; A/B in contiguous k order, not ISA table order):
//   A (16x32, 16-bit): lane row = lane&15; elem i -> K = klo+i,
//                      klo = (lane<16)?0:16
//   B (32x16, 16-bit): lane col = lane&15; elem i -> K = kstart+i,
//                      kstart = (lane<16)?0:16
//   C (16x16, f32):    lane col = lane&15; elem r -> row = mbase+r,
//                      mbase = (lane<16)?0:8
// OUT_MODE: 0 = f16 row-major [S x E]
//           1 = f32 row-major [S x E]   (final output)
//           2 = f16 per-head transposed Vt[b][h][d][key]  (for PV stage)

template <int OUT_MODE>
__global__ __launch_bounds__(128) void gemm_bias(const _Float16* __restrict__ A,
                                                 const _Float16* __restrict__ Wt,
                                                 const float* __restrict__ bias,
                                                 void* __restrict__ out) {
    const int lane = threadIdx.x & 31;
    const int gw   = (blockIdx.x * blockDim.x + threadIdx.x) >> 5;
    const int tilesN = EMB / 64;              // 12
    const int tm = gw / tilesN;
    const int tn = gw % tilesN;
    const int row0 = tm * 16, col0 = tn * 64;

    const int mrow    = lane & 15;
    const int klo     = (lane < 16) ? 0 : 16;
    const int kstartB = (lane < 16) ? 0 : 16;
    const int mbase   = (lane < 16) ? 0 : 8;
    const int ncol    = lane & 15;

    v8f acc[4];
#pragma unroll
    for (int t = 0; t < 4; ++t) acc[t] = (v8f){0, 0, 0, 0, 0, 0, 0, 0};

    const _Float16* arow = A + (size_t)(row0 + mrow) * EMB;
    const _Float16* wrow0 = Wt + (size_t)(col0 + ncol) * EMB;          // +t*16*EMB per tile

    for (int k0 = 0; k0 < EMB; k0 += 32) {
        v16h a;
#pragma unroll
        for (int i = 0; i < 8; ++i) {
            a[i]     = arow[k0 + klo + i];
            a[8 + i] = arow[k0 + klo + 8 + i];
        }
#pragma unroll
        for (int t = 0; t < 4; ++t) {
            v16h b;
            const _Float16* wr = wrow0 + (size_t)t * 16 * EMB + k0 + kstartB;
#pragma unroll
            for (int i = 0; i < 16; ++i) b[i] = wr[i];
            acc[t] = __builtin_amdgcn_wmma_f32_16x16x32_f16(false, a, false, b,
                                                            (short)0, acc[t], false, false);
        }
    }

#pragma unroll
    for (int t = 0; t < 4; ++t) {
        const int col = col0 + t * 16 + ncol;
        const float bv = bias[col];
#pragma unroll
        for (int r = 0; r < 8; ++r) {
            float v = acc[t][r] + bv;
            const int srow = row0 + mbase + r;
            if (OUT_MODE == 1) {
                ((float*)out)[(size_t)srow * EMB + col] = v;
            } else if (OUT_MODE == 0) {
                ((_Float16*)out)[(size_t)srow * EMB + col] = (_Float16)v;
            } else {
                const int bb = srow / NSEQ, nn = srow % NSEQ;
                const int hh = col / HDIM,  dd = col % HDIM;
                ((_Float16*)out)[(((size_t)bb * NHEAD + hh) * HDIM + dd) * NSEQ + nn] =
                    (_Float16)v;
            }
        }
    }
}

// ---------- attention: per (b,h), flash-style online softmax --------------
// One wave handles 16 query rows; loops over keys in chunks of 32.
// Q,K row-major [S x E]; V pre-transposed Vt[b][h][d][key] so all fragment
// loads are contiguous b128. att = softmax(q.k^T) * HDIM (faithful quirk).

__global__ __launch_bounds__(128) void attn_kernel(const _Float16* __restrict__ Q,
                                                   const _Float16* __restrict__ K,
                                                   const _Float16* __restrict__ Vt,
                                                   _Float16* __restrict__ O) {
    __shared__ _Float16 pbuf[4][16 * 32];   // per-wave P staging (transpose via LDS)

    const int lane = threadIdx.x & 31;
    const int widb = threadIdx.x >> 5;
    const int gw   = blockIdx.x * 4 + widb;          // 0..4095
    const int qt   = gw % (NSEQ / 16);
    const int bh   = gw / (NSEQ / 16);
    const int h    = bh % NHEAD;
    const int b    = bh / NHEAD;

    const size_t rowbase = (size_t)b * NSEQ;
    const size_t colbase = (size_t)h * HDIM;

    const int mrow    = lane & 15;
    const int klo     = (lane < 16) ? 0 : 16;
    const int kstartB = (lane < 16) ? 0 : 16;
    const int mbase   = (lane < 16) ? 0 : 8;
    const int ncol    = lane & 15;

    // Q fragments for this query tile (D=96 -> 3 K-chunks of 32)
    v16h qa[3];
    {
        const _Float16* qrow = Q + (rowbase + qt * 16 + mrow) * EMB + colbase;
#pragma unroll
        for (int c = 0; c < 3; ++c) {
#pragma unroll
            for (int i = 0; i < 8; ++i) {
                qa[c][i]     = qrow[c * 32 + klo + i];
                qa[c][8 + i] = qrow[c * 32 + klo + 8 + i];
            }
        }
    }

    float runM[8], runL[8];
    v8f acc[6];
#pragma unroll
    for (int r = 0; r < 8; ++r) { runM[r] = -3.0e38f; runL[r] = 0.0f; }
#pragma unroll
    for (int t = 0; t < 6; ++t) acc[t] = (v8f){0, 0, 0, 0, 0, 0, 0, 0};

    _Float16* pb = &pbuf[widb][0];
    const _Float16* vhead = Vt + ((size_t)b * NHEAD + h) * HDIM * NSEQ;

    for (int j0 = 0; j0 < NSEQ; j0 += 32) {
        // --- energy tiles for 32 keys: E0 (keys j0..j0+15), E1 (j0+16..j0+31)
        v8f e0 = {0, 0, 0, 0, 0, 0, 0, 0};
        v8f e1 = {0, 0, 0, 0, 0, 0, 0, 0};
        {
            const _Float16* krow0 = K + (rowbase + j0 + ncol) * EMB + colbase;
            const _Float16* krow1 = krow0 + (size_t)16 * EMB;
#pragma unroll
            for (int c = 0; c < 3; ++c) {
                v16h b0, b1;
#pragma unroll
                for (int i = 0; i < 16; ++i) {
                    b0[i] = krow0[c * 32 + kstartB + i];
                    b1[i] = krow1[c * 32 + kstartB + i];
                }
                e0 = __builtin_amdgcn_wmma_f32_16x16x32_f16(false, qa[c], false, b0,
                                                            (short)0, e0, false, false);
                e1 = __builtin_amdgcn_wmma_f32_16x16x32_f16(false, qa[c], false, b1,
                                                            (short)0, e1, false, false);
            }
        }

        // --- online softmax bookkeeping (row = mbase + r, reduce across n-lanes)
        float p0[8], p1[8];
#pragma unroll
        for (int r = 0; r < 8; ++r) {
            float mx    = red_max16(fmaxf(e0[r], e1[r]));
            float newM  = fmaxf(runM[r], mx);
            float scale = __expf(runM[r] - newM);
            p0[r] = __expf(e0[r] - newM);
            p1[r] = __expf(e1[r] - newM);
            float rs = red_sum16(p0[r] + p1[r]);
            runL[r] = runL[r] * scale + rs;
            runM[r] = newM;
#pragma unroll
            for (int t = 0; t < 6; ++t) acc[t][r] *= scale;
            // stage P (C-layout -> LDS row-major [16 q rows][32 keys])
            pb[(mbase + r) * 32 + ncol]      = (_Float16)p0[r];
            pb[(mbase + r) * 32 + 16 + ncol] = (_Float16)p1[r];
        }
        __builtin_amdgcn_wave_barrier();   // keep DS store->load order (same wave, in-order DS)

        // --- P as A-fragment (16 x 32)
        v16h pa;
        {
            const _Float16* prow = pb + mrow * 32;
#pragma unroll
            for (int i = 0; i < 8; ++i) {
                pa[i]     = prow[klo + i];
                pa[8 + i] = prow[klo + 8 + i];
            }
        }
        __builtin_amdgcn_wave_barrier();

        // --- O += P (16x32) x V (32x96), 6 column tiles of 16; Vt is [d][key]
#pragma unroll
        for (int t = 0; t < 6; ++t) {
            v16h vb;
            const _Float16* vcol = vhead + (size_t)(t * 16 + ncol) * NSEQ + j0 + kstartB;
#pragma unroll
            for (int i = 0; i < 16; ++i) vb[i] = vcol[i];
            acc[t] = __builtin_amdgcn_wmma_f32_16x16x32_f16(false, pa, false, vb,
                                                            (short)0, acc[t], false, false);
        }
    }

    // --- epilogue: out = acc * (HDIM / L)   (softmax * 96 quirk)
#pragma unroll
    for (int r = 0; r < 8; ++r) {
        float f = 96.0f / runL[r];
        size_t orow = (rowbase + qt * 16 + mbase + r) * EMB + colbase;
#pragma unroll
        for (int t = 0; t < 6; ++t)
            O[orow + t * 16 + ncol] = (_Float16)(acc[t][r] * f);
    }
}

// ---------- host launcher -------------------------------------------------

extern "C" void kernel_launch(void* const* d_in, const int* in_sizes, int n_in,
                              void* d_out, int out_size, void* d_ws, size_t ws_size,
                              hipStream_t stream) {
    const float* x  = (const float*)d_in[0];
    const float* Wq = (const float*)d_in[1];
    const float* bq = (const float*)d_in[2];
    const float* Wk = (const float*)d_in[3];
    const float* bk = (const float*)d_in[4];
    const float* Wv = (const float*)d_in[5];
    const float* bv = (const float*)d_in[6];
    const float* Wp = (const float*)d_in[7];
    const float* bp = (const float*)d_in[8];
    float* out = (float*)d_out;

    const size_t SE = (size_t)SROWS * EMB;   // 6,291,456
    const size_t EE = (size_t)EMB * EMB;     //   589,824

    _Float16* ws  = (_Float16*)d_ws;
    _Float16* xh  = ws;                 // x, f16, row-major
    _Float16* Wqt = xh  + SE;           // weights, f16, TRANSPOSED
    _Float16* Wkt = Wqt + EE;
    _Float16* Wvt = Wkt + EE;
    _Float16* Wpt = Wvt + EE;
    _Float16* Qh  = Wpt + EE;           // Q, f16, row-major [S x E]
    _Float16* Kh  = Qh  + SE;           // K, f16, row-major [S x E]
    _Float16* Vth = Kh  + SE;           // V, f16, transposed Vt[b][h][d][key]
    _Float16* Ah  = Vth + SE;           // attention out, f16, row-major [S x E]

    // 1) fp32 -> f16 (x straight; weights transposed)
    cvt_f32_f16<<<(int)((SE + 255) / 256), 256, 0, stream>>>(x, xh, (int)SE);
    cvt_w_f16t<<<(int)((EE + 255) / 256), 256, 0, stream>>>(Wq, Wqt);
    cvt_w_f16t<<<(int)((EE + 255) / 256), 256, 0, stream>>>(Wk, Wkt);
    cvt_w_f16t<<<(int)((EE + 255) / 256), 256, 0, stream>>>(Wv, Wvt);
    cvt_w_f16t<<<(int)((EE + 255) / 256), 256, 0, stream>>>(Wp, Wpt);

    // 2) Q/K/V projections (WMMA, 16x64 tile per wave)
    const int gemm_blocks = (SROWS / 16) * (EMB / 64) / 4;   // 1536, 4 waves/block
    gemm_bias<0><<<gemm_blocks, 128, 0, stream>>>(xh, Wqt, bq, (void*)Qh);
    gemm_bias<0><<<gemm_blocks, 128, 0, stream>>>(xh, Wkt, bk, (void*)Kh);
    gemm_bias<2><<<gemm_blocks, 128, 0, stream>>>(xh, Wvt, bv, (void*)Vth);

    // 3) attention (WMMA + online softmax)
    const int attn_blocks = BATCH * NHEAD * (NSEQ / 16) / 4; // 1024
    attn_kernel<<<attn_blocks, 128, 0, stream>>>(Qh, Kh, Vth, Ah);

    // 4) output projection, fp32 out (WMMA)
    gemm_bias<1><<<gemm_blocks, 128, 0, stream>>>(Ah, Wpt, bp, (void*)out);
}
